// GAT_CITESEER_50697793962427
// MI455X (gfx1250) — compile-verified, matrix operand order checked by reading, not run on hardware
//
#include <hip/hip_runtime.h>
#include <math.h>

typedef __attribute__((ext_vector_type(16))) __bf16 bf16x16;
typedef __attribute__((ext_vector_type(8)))  float  floatx8;

#define NEG_SLOPE 0.2f

// ---- monotone float <-> ordered-uint encoding (for atomic float max) ----
__device__ __forceinline__ unsigned f2ord(float f) {
  unsigned u = __float_as_uint(f);
  return (u & 0x80000000u) ? ~u : (u | 0x80000000u);
}
__device__ __forceinline__ float ord2f(unsigned u) {
  return __uint_as_float((u & 0x80000000u) ? (u & 0x7FFFFFFFu) : ~u);
}

// ---------------- fills ----------------
__global__ void fill_f32_kernel(float* __restrict__ p, float v, long long n) {
  long long i = (long long)blockIdx.x * blockDim.x + threadIdx.x;
  if (i < n) p[i] = v;
}
__global__ void fill_u32_kernel(unsigned* __restrict__ p, unsigned v, long long n) {
  long long i = (long long)blockIdx.x * blockDim.x + threadIdx.x;
  if (i < n) p[i] = v;
}

// =====================================================================================
// Wide-N bf16 WMMA GEMM:  C[M,NCOLS] = A[M,K] @ B[K,NCOLS],  NCOLS compile-time (200).
// Block = 256 threads = 8 waves; wave owns 16 rows x ALL NT=ceil(NCOLS/16) column
// tiles (13 accumulators in VGPRs) so every A cacheline is read from HBM exactly
// once.  B k-tile staged via registers -> LDS (ping-pong, 1 barrier/iter): the
// next tile's global loads are issued right after the barrier and only hit a
// wait at the *next* iteration's LDS store, so HBM latency overlaps the math.
// =====================================================================================
template <int NCOLS>
__global__ __launch_bounds__(256) void gemm_bf16_wmma_wide_kernel(
    const float* __restrict__ A, const float* __restrict__ B,
    float* __restrict__ C, int M, int K)
{
  constexpr int NT   = (NCOLS + 15) / 16;        // 13 column tiles
  constexpr int SCNT = (NCOLS * 32) / 256;       // 25 staged B elems per thread
  static_assert((NCOLS * 32) % 256 == 0, "staging must divide evenly");

  __shared__ __bf16 Bs[2][NT * 16][36];          // [buf][n][k], 72B row stride

  const int lane = threadIdx.x & 31;
  const int wave = threadIdx.x >> 5;
  const int half = lane >> 4;
  const int l16  = lane & 15;
  const int mBase = blockIdx.x * 128 + wave * 16;
  const int mrow  = mBase + l16;
  const bool mvalid = (mrow < M);
  const float* __restrict__ arow = A + (size_t)mrow * K;

  // zero the N-pad region (cols NCOLS..NT*16) once; staging never touches it
  for (int i = NCOLS * 32 + threadIdx.x; i < NT * 16 * 32; i += 256) {
    Bs[0][i >> 5][i & 31] = (__bf16)0.f;
    Bs[1][i >> 5][i & 31] = (__bf16)0.f;
  }

  // ---- B k-tile -> registers (branchless fast path; guarded K-tail) ----
  auto loadB = [&](int kt, float* r) {
    if (kt + 32 <= K) {
#pragma unroll
      for (int j = 0; j < SCNT; ++j) {
        int i = threadIdx.x + j * 256;
        r[j] = B[(size_t)(kt + (i & 31)) * NCOLS + (i >> 5)];
      }
    } else {
#pragma unroll
      for (int j = 0; j < SCNT; ++j) {
        int i = threadIdx.x + j * 256;
        r[j] = ((kt + (i & 31)) < K)
                   ? B[(size_t)(kt + (i & 31)) * NCOLS + (i >> 5)] : 0.f;
      }
    }
  };
  // ---- registers -> LDS (transposed [n][k]) ----
  auto storeB = [&](const float* r, int bufi) {
#pragma unroll
    for (int j = 0; j < SCNT; ++j) {
      int i = threadIdx.x + j * 256;
      Bs[bufi][i >> 5][i & 31] = (__bf16)r[j];
    }
  };

  floatx8 acc[NT];
  const floatx8 z8 = {0.f, 0.f, 0.f, 0.f, 0.f, 0.f, 0.f, 0.f};
#pragma unroll
  for (int t = 0; t < NT; ++t) acc[t] = z8;

  float breg[SCNT];
  loadB(0, breg);                                // prologue: tile 0 into regs
  int cur = 0;

  for (int k0 = 0; k0 < K; k0 += 32) {
    storeB(breg, cur);                           // regs -> LDS (waits on loads here)
    __syncthreads();
    if (k0 + 32 < K) loadB(k0 + 32, breg);       // prefetch next tile (no wait yet)

    // ---- A fragment straight from global (ISA 16-bit A 16x32 layout) ----
    bf16x16 afrag;
    const int kb = k0 + half * 8;
    if ((k0 + 32 <= K) & (int)mvalid) {
#pragma unroll
      for (int i = 0; i < 8; ++i) {
        afrag[i]     = (__bf16)arow[kb + i];
        afrag[i + 8] = (__bf16)arow[kb + 16 + i];
      }
    } else {
#pragma unroll
      for (int i = 0; i < 8; ++i) {
        int k1 = kb + i, k2 = kb + 16 + i;
        afrag[i]     = (__bf16)((mvalid && k1 < K) ? arow[k1] : 0.f);
        afrag[i + 8] = (__bf16)((mvalid && k2 < K) ? arow[k2] : 0.f);
      }
    }

    // ---- one WMMA per column tile, A fragment reused 13x ----
#pragma unroll
    for (int t = 0; t < NT; ++t) {
      bf16x16 bfrag;
      unsigned* bp = (unsigned*)&bfrag;
      const unsigned* s0 = (const unsigned*)&Bs[cur][t * 16 + l16][half * 8];
      const unsigned* s1 = (const unsigned*)&Bs[cur][t * 16 + l16][16 + half * 8];
#pragma unroll
      for (int i = 0; i < 4; ++i) { bp[i] = s0[i]; bp[4 + i] = s1[i]; }
      acc[t] = __builtin_amdgcn_wmma_f32_16x16x32_bf16(
          false, afrag, false, bfrag, (short)0, acc[t], false, false);
    }
    cur ^= 1;
  }

  // ---- stores: lane holds column l16 of each tile; VGPR j = row half*8+j ----
#pragma unroll
  for (int t = 0; t < NT; ++t) {
    int ncol = t * 16 + l16;
    if (ncol < NCOLS) {
#pragma unroll
      for (int j = 0; j < 8; ++j) {
        int m = mBase + half * 8 + j;
        if (m < M) C[(size_t)m * NCOLS + ncol] = acc[t][j];
      }
    }
  }
}

// ---------------- generic fallback (any Ncols): one 16x16 tile per wave ----------------
__global__ __launch_bounds__(256) void gemm_bf16_wmma_kernel(
    const float* __restrict__ A, const float* __restrict__ B,
    float* __restrict__ C, int M, int K, int Ncols)
{
  __shared__ __bf16 Bs[2][16][36];
  const int lane = threadIdx.x & 31;
  const int wave = threadIdx.x >> 5;
  const int half = lane >> 4;
  const int l16  = lane & 15;
  const int mBase = blockIdx.x * 128 + wave * 16;
  const int nBase = blockIdx.y * 16;
  const int mrow  = mBase + l16;
  const bool mvalid = (mrow < M);
  const bool nfull  = (nBase + 16 <= Ncols);
  const float* __restrict__ arow = A + (size_t)mrow * K;

  const int sk  = threadIdx.x & 31;
  const int sn0 = threadIdx.x >> 5;
  const bool sn0v = (nBase + sn0) < Ncols;
  const bool sn1v = (nBase + sn0 + 8) < Ncols;

  auto stageB = [&](int kt, int bufi) {
    if ((kt + 32 <= K) & (int)nfull) {
      float v0 = B[(size_t)(kt + sk) * Ncols + (nBase + sn0)];
      float v1 = B[(size_t)(kt + sk) * Ncols + (nBase + sn0 + 8)];
      Bs[bufi][sn0][sk]     = (__bf16)v0;
      Bs[bufi][sn0 + 8][sk] = (__bf16)v1;
    } else {
      bool kv = (kt + sk) < K;
      float v0 = (kv && sn0v) ? B[(size_t)(kt + sk) * Ncols + (nBase + sn0)]     : 0.f;
      float v1 = (kv && sn1v) ? B[(size_t)(kt + sk) * Ncols + (nBase + sn0 + 8)] : 0.f;
      Bs[bufi][sn0][sk]     = (__bf16)v0;
      Bs[bufi][sn0 + 8][sk] = (__bf16)v1;
    }
  };

  floatx8 acc = {0.f, 0.f, 0.f, 0.f, 0.f, 0.f, 0.f, 0.f};
  stageB(0, 0);
  int cur = 0;

  for (int k0 = 0; k0 < K; k0 += 32) {
    __syncthreads();
    if (k0 + 32 < K) stageB(k0 + 32, cur ^ 1);

    bf16x16 afrag;
    const int kb = k0 + half * 8;
    if ((k0 + 32 <= K) & (int)mvalid) {
#pragma unroll
      for (int i = 0; i < 8; ++i) {
        afrag[i]     = (__bf16)arow[kb + i];
        afrag[i + 8] = (__bf16)arow[kb + 16 + i];
      }
    } else {
#pragma unroll
      for (int i = 0; i < 8; ++i) {
        int k1 = kb + i, k2 = kb + 16 + i;
        afrag[i]     = (__bf16)((mvalid && k1 < K) ? arow[k1] : 0.f);
        afrag[i + 8] = (__bf16)((mvalid && k2 < K) ? arow[k2] : 0.f);
      }
    }

    bf16x16 bfrag;
    {
      unsigned* bp = (unsigned*)&bfrag;
      const unsigned* s0 = (const unsigned*)&Bs[cur][l16][half * 8];
      const unsigned* s1 = (const unsigned*)&Bs[cur][l16][16 + half * 8];
#pragma unroll
      for (int i = 0; i < 4; ++i) { bp[i] = s0[i]; bp[4 + i] = s1[i]; }
    }

    acc = __builtin_amdgcn_wmma_f32_16x16x32_bf16(
        false, afrag, false, bfrag, (short)0, acc, false, false);
    cur ^= 1;
  }

  const int ncol = nBase + l16;
  if (ncol < Ncols) {
#pragma unroll
    for (int j = 0; j < 8; ++j) {
      int m = mBase + half * 8 + j;
      if (m < M) C[(size_t)m * Ncols + ncol] = acc[j];
    }
  }
}

// ---------------- per-node attention coefficients: out[n,h] = <h[n,h,:], att[h,:]> ----------------
__global__ void attn_coef_kernel(const float* __restrict__ h, const float* __restrict__ att,
                                 float* __restrict__ out, int N, int H, int F) {
  int idx = blockIdx.x * blockDim.x + threadIdx.x;
  if (idx >= N * H) return;
  int n = idx / H, hh = idx - n * H;
  const float* hp = h + ((size_t)n * H + hh) * F;
  const float* ap = att + (size_t)hh * F;
  float s = 0.f;
  for (int f = 0; f < F; ++f) s = fmaf(hp[f], ap[f], s);
  out[idx] = s;
}

// ---------------- pass 1: edge logits + segment max (atomic on ordered uint) ----------------
__global__ void edge_logits_kernel(const int* __restrict__ src, const int* __restrict__ dst,
                                   int E, int N, int H,
                                   const float* __restrict__ asrc, const float* __restrict__ adst,
                                   float* __restrict__ ebuf, unsigned* __restrict__ mmax) {
  long long idx = (long long)blockIdx.x * blockDim.x + threadIdx.x;
  long long total = (long long)(E + N) * H;
  if (idx >= total) return;
  int hh = (int)(idx % H);
  int e  = (int)(idx / H);
  int s = (e < E) ? src[e] : (e - E);
  int d = (e < E) ? dst[e] : (e - E);
  float v = asrc[(size_t)s * H + hh] + adst[(size_t)d * H + hh];
  v = (v > 0.f) ? v : NEG_SLOPE * v;
  ebuf[idx] = v;
  atomicMax(&mmax[(size_t)d * H + hh], f2ord(v));
}

// ---------------- pass 2: p = exp(e - m[dst]); denom += p ----------------
__global__ void edge_exp_kernel(const int* __restrict__ src, const int* __restrict__ dst,
                                int E, int N, int H,
                                float* __restrict__ ebuf, const unsigned* __restrict__ mmax,
                                float* __restrict__ den) {
  long long idx = (long long)blockIdx.x * blockDim.x + threadIdx.x;
  long long total = (long long)(E + N) * H;
  if (idx >= total) return;
  int hh = (int)(idx % H);
  int e  = (int)(idx / H);
  int d = (e < E) ? dst[e] : (e - E);
  float m = ord2f(mmax[(size_t)d * H + hh]);
  float p = __expf(ebuf[idx] - m);
  ebuf[idx] = p;
  atomicAdd(&den[(size_t)d * H + hh], p);
}

// ---------------- pass 3: out[dst,h,f] += (p/den[dst,h]) * h[src,h,f] ----------------
__global__ void edge_aggr_kernel(const int* __restrict__ src, const int* __restrict__ dst,
                                 int E, int N, int H, int F,
                                 const float* __restrict__ ebuf, const float* __restrict__ den,
                                 const float* __restrict__ h, float* __restrict__ out) {
  long long idx = (long long)blockIdx.x * blockDim.x + threadIdx.x;
  long long total = (long long)(E + N) * H * F;
  if (idx >= total) return;
  int f = (int)(idx % F);
  long long t = idx / F;
  int hh = (int)(t % H);
  int e  = (int)(t / H);
  int s = (e < E) ? src[e] : (e - E);
  int d = (e < E) ? dst[e] : (e - E);
  float alpha = ebuf[(size_t)e * H + hh] / (den[(size_t)d * H + hh] + 1e-16f);
  atomicAdd(&out[((size_t)d * H + hh) * F + f],
            alpha * h[((size_t)s * H + hh) * F + f]);
}

// ---------------- bias + ELU (in place) ----------------
__global__ void bias_elu_kernel(float* __restrict__ x, const float* __restrict__ b,
                                long long total, int D) {
  long long idx = (long long)blockIdx.x * blockDim.x + threadIdx.x;
  if (idx >= total) return;
  int j = (int)(idx % D);
  float v = x[idx] + b[j];
  x[idx] = (v > 0.f) ? v : (__expf(v) - 1.f);
}

// ---------------- final linear + log_softmax ----------------
__global__ void head_kernel(const float* __restrict__ h, const float* __restrict__ Wl,
                            const float* __restrict__ bl, float* __restrict__ out,
                            int N, int D, int C) {
  int n = blockIdx.x * blockDim.x + threadIdx.x;
  if (n >= N) return;
  float acc[8];
  for (int c = 0; c < C; ++c) acc[c] = bl[c];
  const float* hp = h + (size_t)n * D;
  for (int k = 0; k < D; ++k) {
    float hv = hp[k];
    const float* wr = Wl + (size_t)k * C;
    for (int c = 0; c < C; ++c) acc[c] = fmaf(hv, wr[c], acc[c]);
  }
  float mx = acc[0];
  for (int c = 1; c < C; ++c) mx = fmaxf(mx, acc[c]);
  float s = 0.f;
  for (int c = 0; c < C; ++c) s += __expf(acc[c] - mx);
  float ls = __logf(s);
  for (int c = 0; c < C; ++c) out[(size_t)n * C + c] = acc[c] - mx - ls;
}

// =================================================================================
extern "C" void kernel_launch(void* const* d_in, const int* in_sizes, int n_in,
                              void* d_out, int out_size, void* d_ws, size_t ws_size,
                              hipStream_t stream) {
  (void)n_in; (void)out_size; (void)ws_size;
  const float* x   = (const float*)d_in[0];
  const int*   ei  = (const int*)d_in[1];
  const float* W1  = (const float*)d_in[2];
  const float* as1 = (const float*)d_in[3];
  const float* ad1 = (const float*)d_in[4];
  const float* b1  = (const float*)d_in[5];
  const float* W2  = (const float*)d_in[6];
  const float* as2 = (const float*)d_in[7];
  const float* ad2 = (const float*)d_in[8];
  const float* b2  = (const float*)d_in[9];
  const float* Wl  = (const float*)d_in[10];
  const float* bl  = (const float*)d_in[11];
  float* out = (float*)d_out;

  const int E    = in_sizes[1] / 2;
  const int D1   = in_sizes[5];              // 200 = H1*HID
  const int FIN  = in_sizes[2] / D1;         // 3703
  const int N    = in_sizes[0] / FIN;        // 50000
  const int H1   = 2;
  const int F1   = D1 / H1;                  // 100
  const int Ccls = in_sizes[11];             // 6
  const int EP   = E + N;

  const int* srcI = ei;
  const int* dstI = ei + E;

  float* ws = (float*)d_ws;
  size_t off = 0;
  float* h1   = ws + off; off += (size_t)N * D1;
  float* hbuf = ws + off; off += (size_t)N * D1;
  float* obuf = ws + off; off += (size_t)N * D1;
  float* ebuf = ws + off; off += (size_t)EP * H1;
  float* asrc = ws + off; off += (size_t)N * H1;
  float* adst = ws + off; off += (size_t)N * H1;
  float* den  = ws + off; off += (size_t)N * H1;
  unsigned* mmax = (unsigned*)(ws + off); off += (size_t)N * H1;

  auto cdiv = [](long long a, long long b) { return (int)((a + b - 1) / b); };
  const int TB = 256;

  auto gemm = [&](const float* A, const float* B, float* C, int M, int K, int Ncols) {
    if (Ncols == 200) {
      dim3 g(cdiv(M, 128), 1);
      gemm_bf16_wmma_wide_kernel<200><<<g, TB, 0, stream>>>(A, B, C, M, K);
    } else {
      dim3 g(cdiv(M, 128), cdiv(Ncols, 16));
      gemm_bf16_wmma_kernel<<<g, TB, 0, stream>>>(A, B, C, M, K, Ncols);
    }
  };

  // ================= layer 1 =================
  fill_f32_kernel<<<cdiv((long long)N * D1, TB), TB, 0, stream>>>(hbuf, 0.f, (long long)N * D1);
  fill_f32_kernel<<<cdiv((long long)N * H1, TB), TB, 0, stream>>>(den, 0.f, (long long)N * H1);
  fill_u32_kernel<<<cdiv((long long)N * H1, TB), TB, 0, stream>>>(mmax, 0u, (long long)N * H1);

  gemm(x, W1, h1, N, FIN, D1);

  attn_coef_kernel<<<cdiv((long long)N * H1, TB), TB, 0, stream>>>(h1, as1, asrc, N, H1, F1);
  attn_coef_kernel<<<cdiv((long long)N * H1, TB), TB, 0, stream>>>(h1, ad1, adst, N, H1, F1);

  edge_logits_kernel<<<cdiv((long long)EP * H1, TB), TB, 0, stream>>>(
      srcI, dstI, E, N, H1, asrc, adst, ebuf, mmax);
  edge_exp_kernel<<<cdiv((long long)EP * H1, TB), TB, 0, stream>>>(
      srcI, dstI, E, N, H1, ebuf, mmax, den);
  edge_aggr_kernel<<<cdiv((long long)EP * H1 * F1, TB), TB, 0, stream>>>(
      srcI, dstI, E, N, H1, F1, ebuf, den, h1, hbuf);
  bias_elu_kernel<<<cdiv((long long)N * D1, TB), TB, 0, stream>>>(hbuf, b1, (long long)N * D1, D1);

  // ================= layer 2 =================
  fill_f32_kernel<<<cdiv((long long)N * D1, TB), TB, 0, stream>>>(obuf, 0.f, (long long)N * D1);
  fill_f32_kernel<<<cdiv((long long)N, TB), TB, 0, stream>>>(den, 0.f, (long long)N);
  fill_u32_kernel<<<cdiv((long long)N, TB), TB, 0, stream>>>(mmax, 0u, (long long)N);

  gemm(hbuf, W2, h1, N, D1, D1);

  attn_coef_kernel<<<cdiv((long long)N, TB), TB, 0, stream>>>(h1, as2, asrc, N, 1, D1);
  attn_coef_kernel<<<cdiv((long long)N, TB), TB, 0, stream>>>(h1, ad2, adst, N, 1, D1);

  edge_logits_kernel<<<cdiv((long long)EP, TB), TB, 0, stream>>>(
      srcI, dstI, E, N, 1, asrc, adst, ebuf, mmax);
  edge_exp_kernel<<<cdiv((long long)EP, TB), TB, 0, stream>>>(
      srcI, dstI, E, N, 1, ebuf, mmax, den);
  edge_aggr_kernel<<<cdiv((long long)EP * D1, TB), TB, 0, stream>>>(
      srcI, dstI, E, N, 1, D1, ebuf, den, h1, obuf);
  bias_elu_kernel<<<cdiv((long long)N * D1, TB), TB, 0, stream>>>(obuf, b2, (long long)N * D1, D1);

  // ================= head =================
  head_kernel<<<cdiv((long long)N, TB), TB, 0, stream>>>(obuf, Wl, bl, out, N, D1, Ccls);
}
